// DisentangledSlotAttention_45672682225930
// MI455X (gfx1250) — matrix operand order repaired and checked by reading, not compiled, not run on hardware
//
#include <hip/hip_runtime.h>

#define B_SZ    32
#define N_TOK   4096
#define D_IN    768
#define D_SLOT  256
#define K_SLOT  8
#define D_SH    128

#define MT      32      // row tile of fused projection kernel
#define LDST    40      // bf16 LDS stride for 32-K staged tiles (pad vs bank conflicts)
#define LDC     260     // f32 LDS stride for LN tile
#define LDN     264     // bf16 LDS stride for normalized tile / q tile

typedef __attribute__((ext_vector_type(16))) __bf16 bf16x16;
typedef __attribute__((ext_vector_type(8)))  __bf16 bf16x8;
typedef __attribute__((ext_vector_type(2)))  __bf16 bf16x2;
typedef __attribute__((ext_vector_type(8)))  float  floatx8;

__device__ __forceinline__ floatx8 wmma_bf16(bf16x16 a, bf16x16 b, floatx8 c) {
  return __builtin_amdgcn_wmma_f32_16x16x32_bf16(false, a, false, b, (short)0, c,
                                                 false, false);
}

// pack two floats into adjacent bf16 (maps onto v_cvt_pk_bf16_f32)
__device__ __forceinline__ bf16x2 pk2(float a, float b) {
  bf16x2 r; r[0] = (__bf16)a; r[1] = (__bf16)b; return r;
}

// A fragment: 16x32 tile stored row-major [row][k] with stride ld (elements).
// lane&15 = row; contiguous 16-K run starting at (lane>>4)*16.
__device__ __forceinline__ bf16x16 frag_a(const __bf16* tile, int ld) {
  const int lane = threadIdx.x & 31;
  const __bf16* p = tile + (lane & 15) * ld + ((lane >> 4) << 4);
  bf16x16 v;
  *((bf16x8*)&v)     = *((const bf16x8*)p);
  *((bf16x8*)&v + 1) = *((const bf16x8*)(p + 8));
  return v;
}

// B fragment: 32(K)x16(N) tile staged as [col][k] with stride ld (elements).
// lane&15 = col; contiguous 16-K run starting at (lane>>4)*16.
__device__ __forceinline__ bf16x16 frag_b(const __bf16* tile, int ld) {
  const int lane = threadIdx.x & 31;
  const __bf16* p = tile + (lane & 15) * ld + ((lane >> 4) << 4);
  bf16x16 v;
  *((bf16x8*)&v)     = *((const bf16x8*)p);
  *((bf16x8*)&v + 1) = *((const bf16x8*)(p + 8));
  return v;
}

// ---------------------------------------------------------------------------
// Kernel 1: inputs = LN(features @ Wp + bp); k_s = inp_s @ Wks; k_t = inp_t @ Wkt
// Also emits per-block column-sum partials of features for the slot-init pooling
// (features are read exactly once from HBM, with b128 loads).
// Grid: (B*N)/MT blocks of 256 threads (8 waves).
// ---------------------------------------------------------------------------
__global__ __launch_bounds__(256) void k_proj_ln_keys(
    const float* __restrict__ features, const float* __restrict__ Wp,
    const float* __restrict__ bp, const float* __restrict__ g,
    const float* __restrict__ bln, const float* __restrict__ Wks,
    const float* __restrict__ Wkt, float* __restrict__ inputs,
    float* __restrict__ ks_out, float* __restrict__ kt_out,
    float* __restrict__ featp)
{
  __shared__ __align__(16) char smem[33280 + 3072 + 16896 + 4096];
  __bf16* sA   = (__bf16*)smem;                 // MT x LDST (2560 B)
  __bf16* sB   = (__bf16*)(smem + 2560);        // 256 x LDST (20480 B)
  float*  sOut = (float*)smem;                  // MT x LDC f32 (aliases staging)
  float*  sCol = (float*)(smem + 33280);        // 768 column sums / LN stats
  __bf16* sN   = (__bf16*)(smem + 36352);       // MT x LDN normalized bf16
  float*  sCp  = (float*)(smem + 53248);        // 256 threads x 4 pooling partials
  __bf16* sW2  = (__bf16*)smem;                 // 128 x LDST (stage 3, aliases)

  const int tid  = threadIdx.x;
  const int wave = tid >> 5, lane = tid & 31;
  const int row0 = blockIdx.x * MT;

  floatx8 acc[2][2];
#pragma unroll
  for (int m = 0; m < 2; ++m)
#pragma unroll
    for (int c = 0; c < 2; ++c) acc[m][c] = (floatx8)0.f;

  // per-thread b128 staging coordinates
  const int arow = tid >> 3, akq = (tid & 7) << 2;       // features tile
  for (int kc = 0; kc < D_IN / 32; ++kc) {
    const int k0 = kc * 32;
    // --- features 32x32 chunk: one float4 per thread ---
    {
      const float4 av = *(const float4*)(features +
          (size_t)(row0 + arow) * D_IN + k0 + akq);
      __bf16* d = sA + arow * LDST + akq;
      *(bf16x2*)(d + 0) = pk2(av.x, av.y);
      *(bf16x2*)(d + 2) = pk2(av.z, av.w);
      float* cp = sCp + tid * 4;
      cp[0] = av.x; cp[1] = av.y; cp[2] = av.z; cp[3] = av.w;
    }
    // --- Wp 32x256 chunk, transposed to [col][k] as k-pairs: 4 units/thread ---
#pragma unroll
    for (int i = 0; i < 4; ++i) {
      const int q = tid + i * 256;              // 1024 units (16 k-pairs x 64 colq)
      const int kp2 = (q >> 6) << 1;            // even k row
      const int colq = (q & 63) << 2;
      const float4 w0 = *(const float4*)(Wp + (size_t)(k0 + kp2) * D_SLOT + colq);
      const float4 w1 = *(const float4*)(Wp + (size_t)(k0 + kp2 + 1) * D_SLOT + colq);
      *(bf16x2*)(sB + (colq + 0) * LDST + kp2) = pk2(w0.x, w1.x);
      *(bf16x2*)(sB + (colq + 1) * LDST + kp2) = pk2(w0.y, w1.y);
      *(bf16x2*)(sB + (colq + 2) * LDST + kp2) = pk2(w0.z, w1.z);
      *(bf16x2*)(sB + (colq + 3) * LDST + kp2) = pk2(w0.w, w1.w);
    }
    __syncthreads();
    if (tid < 32) {                 // deterministic pooling column reduction
      const int gq = tid >> 2, comp = tid & 3;
      float s = 0.f;
#pragma unroll
      for (int row = 0; row < 32; ++row) s += sCp[(row * 8 + gq) * 4 + comp];
      sCol[k0 + tid] = s;           // each chunk owns its 32 columns
    }
    bf16x16 bf0 = frag_b(sB + (wave * 2 + 0) * 16 * LDST, LDST);
    bf16x16 bf1 = frag_b(sB + (wave * 2 + 1) * 16 * LDST, LDST);
#pragma unroll
    for (int m = 0; m < 2; ++m) {
      bf16x16 af = frag_a(sA + m * 16 * LDST, LDST);
      acc[m][0] = wmma_bf16(af, bf0, acc[m][0]);
      acc[m][1] = wmma_bf16(af, bf1, acc[m][1]);
    }
    __syncthreads();
  }

  // feature pooling partials (per-block slice, no atomics -> deterministic)
#pragma unroll
  for (int i = 0; i < 3; ++i)
    featp[(size_t)blockIdx.x * D_IN + tid + i * 256] = sCol[tid + i * 256];

  // bias add, dump accumulators to LDS for LayerNorm
#pragma unroll
  for (int m = 0; m < 2; ++m)
#pragma unroll
    for (int c = 0; c < 2; ++c) {
      const int col = (wave * 2 + c) * 16 + (lane & 15);
      const int rb  = m * 16 + ((lane >> 4) << 3);
      const float bcol = bp[col];
#pragma unroll
      for (int r = 0; r < 8; ++r)
        sOut[(rb + r) * LDC + col] = acc[m][c][r] + bcol;
    }
  __syncthreads();

  // LN statistics: 8 threads per row, shuffle-reduced
  {
    const int rr = tid >> 3, sub = tid & 7;
    const float* rp = sOut + rr * LDC + sub * 32;
    float s = 0.f, s2 = 0.f;
#pragma unroll
    for (int j = 0; j < 32; ++j) { float v = rp[j]; s += v; s2 += v * v; }
#pragma unroll
    for (int m = 1; m < 8; m <<= 1) {
      s  += __shfl_xor(s,  m, 32);
      s2 += __shfl_xor(s2, m, 32);
    }
    if (sub == 0) {
      float mean = s * (1.f / 256.f);
      float var  = s2 * (1.f / 256.f) - mean * mean;
      sCol[rr]      = mean;
      sCol[32 + rr] = rsqrtf(var + 1e-5f);
    }
  }
  __syncthreads();

  // apply LN, write inputs, keep a bf16 copy for the key projections
  {
    const float gg = g[tid], bb = bln[tid];
#pragma unroll 4
    for (int r = 0; r < MT; ++r) {
      float v = (sOut[r * LDC + tid] - sCol[r]) * sCol[32 + r] * gg + bb;
      inputs[(size_t)(row0 + r) * D_SLOT + tid] = v;
      sN[r * LDN + tid] = (__bf16)v;
    }
  }
  __syncthreads();

  // k_s / k_t from the normalized tile still resident in LDS
#pragma unroll
  for (int half = 0; half < 2; ++half) {
    const float* W  = half ? Wkt : Wks;
    float* outp     = half ? kt_out : ks_out;
    floatx8 a2[2];
    a2[0] = (floatx8)0.f; a2[1] = (floatx8)0.f;
    for (int kc = 0; kc < 4; ++kc) {
      // stage W^T 32x128 chunk as [col][k] k-pairs: 2 units per thread
#pragma unroll
      for (int i = 0; i < 2; ++i) {
        const int q = tid + i * 256;            // 512 units (16 k-pairs x 32 colq)
        const int kp2 = (q >> 5) << 1;
        const int colq = (q & 31) << 2;
        const float4 w0 = *(const float4*)(W + (size_t)(kc * 32 + kp2) * D_SH + colq);
        const float4 w1 = *(const float4*)(W + (size_t)(kc * 32 + kp2 + 1) * D_SH + colq);
        *(bf16x2*)(sW2 + (colq + 0) * LDST + kp2) = pk2(w0.x, w1.x);
        *(bf16x2*)(sW2 + (colq + 1) * LDST + kp2) = pk2(w0.y, w1.y);
        *(bf16x2*)(sW2 + (colq + 2) * LDST + kp2) = pk2(w0.z, w1.z);
        *(bf16x2*)(sW2 + (colq + 3) * LDST + kp2) = pk2(w0.w, w1.w);
      }
      __syncthreads();
      bf16x16 bf = frag_b(sW2 + wave * 16 * LDST, LDST);
#pragma unroll
      for (int m = 0; m < 2; ++m) {
        bf16x16 af = frag_a(sN + m * 16 * LDN + half * D_SH + kc * 32, LDN);
        a2[m] = wmma_bf16(af, bf, a2[m]);
      }
      __syncthreads();
    }
#pragma unroll
    for (int m = 0; m < 2; ++m) {
      const int col = wave * 16 + (lane & 15);
      const int rb  = m * 16 + ((lane >> 4) << 3);
#pragma unroll
      for (int r = 0; r < 8; ++r)
        outp[(size_t)(row0 + rb + r) * D_SH + col] = a2[m][r];
    }
  }
}

// ---------------------------------------------------------------------------
// Pool reduce: featpool[b][d] = mean over N of features  (from block partials)
// ---------------------------------------------------------------------------
__global__ __launch_bounds__(256) void k_pool_reduce(
    const float* __restrict__ featp, float* __restrict__ featpool)
{
  const int b = blockIdx.x, t = threadIdx.x;
  for (int d = t; d < D_IN; d += 256) {
    float s = 0.f;
    for (int j = 0; j < N_TOK / MT; ++j)
      s += featp[(size_t)(b * (N_TOK / MT) + j) * D_IN + d];
    featpool[b * D_IN + d] = s * (1.f / (float)N_TOK);
  }
}

// ---------------------------------------------------------------------------
// Slot init MLP: slots = gelu(featpool @ Wi1 + bi1) @ Wi2 + bi2
// ---------------------------------------------------------------------------
__global__ __launch_bounds__(256) void k_init_slots(
    const float* __restrict__ featpool, const float* __restrict__ Wi1,
    const float* __restrict__ bi1, const float* __restrict__ Wi2,
    const float* __restrict__ bi2, float* __restrict__ slots)
{
  __shared__ float fp[D_IN];
  __shared__ float hid[2 * D_SLOT];
  const int b = blockIdx.x, t = threadIdx.x;
  for (int i = t; i < D_IN; i += 256) fp[i] = featpool[b * D_IN + i];
  __syncthreads();
#pragma unroll
  for (int c = 0; c < 2; ++c) {
    const int j = t + c * 256;
    float a = bi1[j];
    for (int d = 0; d < D_IN; ++d) a += fp[d] * Wi1[(size_t)d * 512 + j];
    hid[j] = 0.5f * a * (1.f + erff(a * 0.70710678118654752f));
  }
  __syncthreads();
#pragma unroll
  for (int c = 0; c < 8; ++c) {
    const int j = t + c * 256;
    float a = bi2[j];
    for (int d = 0; d < 512; ++d) a += hid[d] * Wi2[(size_t)d * 2048 + j];
    slots[(size_t)b * 2048 + j] = a;
  }
}

// ---------------------------------------------------------------------------
// Q projection with softmax scales folded in; rows 8..15 zero-padded for WMMA.
// ---------------------------------------------------------------------------
__global__ __launch_bounds__(256) void k_qproj(
    const float* __restrict__ slots, const float* __restrict__ Wqs,
    const float* __restrict__ Wqt, float* __restrict__ qs, float* __restrict__ qt)
{
  __shared__ float ss[K_SLOT * D_SLOT];
  const int b = blockIdx.x, t = threadIdx.x;
#pragma unroll
  for (int i = 0; i < 8; ++i) ss[i * 256 + t] = slots[(size_t)(b * 8 + i) * 256 + t];
  __syncthreads();
  const float cs = 0.7f * 0.088388347648318447f;   // 0.7 / sqrt(128)
  const float ct = 0.3f * 0.088388347648318447f;   // 0.3 / sqrt(128)
#pragma unroll
  for (int i = 0; i < 4; ++i) {
    const int idx = t + i * 256;
    const int k = idx >> 7, c = idx & 127;
    float a = 0.f, a2 = 0.f;
    for (int d = 0; d < 128; ++d) {
      a  += ss[k * 256 + d]       * Wqs[d * 128 + c];
      a2 += ss[k * 256 + 128 + d] * Wqt[d * 128 + c];
    }
    qs[(size_t)(b * 16 + k) * 128 + c] = cs * a;
    qt[(size_t)(b * 16 + k) * 128 + c] = ct * a2;
    qs[(size_t)(b * 16 + 8 + k) * 128 + c] = 0.f;
    qt[(size_t)(b * 16 + 8 + k) * 128 + c] = 0.f;
  }
}

// ---------------------------------------------------------------------------
// Attention: logits via WMMA (K=256 concat of shape/texture), softmax over the
// 8 slots (lane-local in the C layout), write attn + per-block rowsum partials.
// Grid: (N/256, B).
// ---------------------------------------------------------------------------
__global__ __launch_bounds__(256) void k_attn(
    const float* __restrict__ qs, const float* __restrict__ qt,
    const float* __restrict__ ks, const float* __restrict__ kt,
    float* __restrict__ attn, float* __restrict__ rspart)
{
  __shared__ __align__(16) __bf16 sQ[16 * LDN];
  __shared__ float wpart[8][2][K_SLOT];
  const int b = blockIdx.y;
  const int n0 = blockIdx.x * 256;
  const int tid = threadIdx.x, wave = tid >> 5, lane = tid & 31;

#pragma unroll
  for (int r = 0; r < 16; ++r) {
    float v = (tid < 128) ? qs[(size_t)(b * 16 + r) * D_SH + tid]
                          : qt[(size_t)(b * 16 + r) * D_SH + (tid - 128)];
    sQ[r * LDN + tid] = (__bf16)v;
  }
  __syncthreads();

  floatx8 acc[2];
  acc[0] = (floatx8)0.f; acc[1] = (floatx8)0.f;
#pragma unroll
  for (int half = 0; half < 2; ++half) {
    const float* kp = half ? kt : ks;
#pragma unroll
    for (int kc = 0; kc < 4; ++kc) {
      const int d0 = kc * 32;
      bf16x16 af = frag_a(sQ + half * D_SH + d0, LDN);
#pragma unroll
      for (int c = 0; c < 2; ++c) {
        const int n = n0 + (wave * 2 + c) * 16 + (lane & 15);
        const float4* p4 = (const float4*)(kp + (size_t)(b * N_TOK + n) * D_SH +
                                           d0 + ((lane >> 4) << 4));
        bf16x16 bf;
        float4 x0 = p4[0], x1 = p4[1], x2 = p4[2], x3 = p4[3];
        *((bf16x2*)&bf + 0) = pk2(x0.x, x0.y);
        *((bf16x2*)&bf + 1) = pk2(x0.z, x0.w);
        *((bf16x2*)&bf + 2) = pk2(x1.x, x1.y);
        *((bf16x2*)&bf + 3) = pk2(x1.z, x1.w);
        *((bf16x2*)&bf + 4) = pk2(x2.x, x2.y);
        *((bf16x2*)&bf + 5) = pk2(x2.z, x2.w);
        *((bf16x2*)&bf + 6) = pk2(x3.x, x3.y);
        *((bf16x2*)&bf + 7) = pk2(x3.z, x3.w);
        acc[c] = wmma_bf16(af, bf, acc[c]);
      }
    }
  }

  // lanes 0..15 hold slots 0..7 in regs 0..7 of their token column
  if (lane < 16) {
#pragma unroll
    for (int c = 0; c < 2; ++c) {
      const int n = n0 + (wave * 2 + c) * 16 + lane;
      float mx = acc[c][0];
#pragma unroll
      for (int r = 1; r < 8; ++r) mx = fmaxf(mx, acc[c][r]);
      float e[8], s = 0.f;
#pragma unroll
      for (int r = 0; r < 8; ++r) { e[r] = __expf(acc[c][r] - mx); s += e[r]; }
      const float inv = 1.f / s;
#pragma unroll
      for (int r = 0; r < 8; ++r) {
        e[r] *= inv;
        attn[(size_t)(b * K_SLOT + r) * N_TOK + n] = e[r];
      }
#pragma unroll
      for (int r = 0; r < 8; ++r) {
        float v = e[r];
#pragma unroll
        for (int m = 1; m < 16; m <<= 1) v += __shfl_xor(v, m, 32);
        if (lane == 0) wpart[wave][c][r] = v;
      }
    }
  }
  __syncthreads();
  if (tid < K_SLOT) {                    // deterministic fixed-order sum
    float s = 0.f;
#pragma unroll
    for (int w = 0; w < 8; ++w)
#pragma unroll
      for (int c = 0; c < 2; ++c) s += wpart[w][c][tid];
    rspart[(size_t)(b * 16 + blockIdx.x) * K_SLOT + tid] = s;
  }
}

__global__ __launch_bounds__(256) void k_rowsum_reduce(
    const float* __restrict__ rspart, float* __restrict__ rowsum)
{
  const int i = threadIdx.x;             // i = b*8 + k  (256 total)
  float s = 0.f;
  for (int j = 0; j < 16; ++j)
    s += rspart[(size_t)((i >> 3) * 16 + j) * K_SLOT + (i & 7)];
  rowsum[i] = s;
}

// ---------------------------------------------------------------------------
// updates = (attn / (rowsum+eps)) @ inputs, 8-way token split, partial outputs.
// Grid: (B, 8).
// ---------------------------------------------------------------------------
__global__ __launch_bounds__(256) void k_updates(
    const float* __restrict__ attn, const float* __restrict__ rowsum,
    const float* __restrict__ inputs, float* __restrict__ updp)
{
  __shared__ __align__(16) __bf16 sA[16 * LDST];
  __shared__ __align__(16) __bf16 sB[256 * LDST];
  __shared__ float sInv[K_SLOT];
  const int b = blockIdx.x, split = blockIdx.y;
  const int tid = threadIdx.x, wave = tid >> 5, lane = tid & 31;

  if (tid < K_SLOT) sInv[tid] = 1.f / (rowsum[b * K_SLOT + tid] + 1e-8f);
  { const int r = 8 + (tid >> 5), k = tid & 31; sA[r * LDST + k] = (__bf16)0.f; }
  __syncthreads();

  floatx8 acc[2];
  acc[0] = (floatx8)0.f; acc[1] = (floatx8)0.f;
  for (int nc = 0; nc < 16; ++nc) {
    const int n0 = split * 512 + nc * 32;
    { const int r = tid >> 5, k = tid & 31;
      sA[r * LDST + k] =
          (__bf16)(attn[(size_t)(b * K_SLOT + r) * N_TOK + n0 + k] * sInv[r]); }
    // stage inputs 32x256 chunk transposed to [col][n] n-pairs: 4 units/thread
#pragma unroll
    for (int i = 0; i < 4; ++i) {
      const int q = tid + i * 256;              // 1024 units (16 n-pairs x 64 colq)
      const int np2 = (q >> 6) << 1;
      const int colq = (q & 63) << 2;
      const float4 v0 = *(const float4*)(inputs +
          (size_t)(b * N_TOK + n0 + np2) * D_SLOT + colq);
      const float4 v1 = *(const float4*)(inputs +
          (size_t)(b * N_TOK + n0 + np2 + 1) * D_SLOT + colq);
      *(bf16x2*)(sB + (colq + 0) * LDST + np2) = pk2(v0.x, v1.x);
      *(bf16x2*)(sB + (colq + 1) * LDST + np2) = pk2(v0.y, v1.y);
      *(bf16x2*)(sB + (colq + 2) * LDST + np2) = pk2(v0.z, v1.z);
      *(bf16x2*)(sB + (colq + 3) * LDST + np2) = pk2(v0.w, v1.w);
    }
    __syncthreads();
    bf16x16 af = frag_a(sA, LDST);
    acc[0] = wmma_bf16(af, frag_b(sB + (wave * 2 + 0) * 16 * LDST, LDST), acc[0]);
    acc[1] = wmma_bf16(af, frag_b(sB + (wave * 2 + 1) * 16 * LDST, LDST), acc[1]);
    __syncthreads();
  }
  if (lane < 16) {
#pragma unroll
    for (int c = 0; c < 2; ++c) {
      const int col = (wave * 2 + c) * 16 + lane;
#pragma unroll
      for (int r = 0; r < 8; ++r)
        updp[((size_t)(b * 8 + split) * K_SLOT + r) * D_SLOT + col] = acc[c][r];
    }
  }
}

__global__ __launch_bounds__(256) void k_upd_reduce(
    const float* __restrict__ updp, float* __restrict__ upd)
{
  const int i = blockIdx.x * 256 + threadIdx.x;  // (b*8+k)*256+col, 65536 total
  const int b = i >> 11, rem = i & 2047;
  float s = 0.f;
  for (int j = 0; j < 8; ++j)
    s += updp[(size_t)(b * 8 + j) * 2048 + rem];
  upd[i] = s;
}

// ---------------------------------------------------------------------------
// GRU cell + LayerNorm + GELU MLP residual per (b,k) row. Grid: B*K rows.
// ---------------------------------------------------------------------------
__global__ __launch_bounds__(256) void k_gru_mlp(
    const float* __restrict__ updates, const float* __restrict__ slots,
    const float* __restrict__ W_ih, const float* __restrict__ W_hh,
    const float* __restrict__ b_ih, const float* __restrict__ b_hh,
    const float* __restrict__ ln_g, const float* __restrict__ ln_b,
    const float* __restrict__ Wm1, const float* __restrict__ bm1,
    const float* __restrict__ Wm2, const float* __restrict__ bm2,
    float* __restrict__ slots_out)
{
  __shared__ float sx[D_SLOT], sh[D_SLOT], shn[D_SLOT], ssn[D_SLOT];
  __shared__ float shid[2 * D_SLOT];
  __shared__ float red[16];
  const int row = blockIdx.x, t = threadIdx.x;
  sx[t] = updates[(size_t)row * D_SLOT + t];
  sh[t] = slots[(size_t)row * D_SLOT + t];
  __syncthreads();

  float gx[3], gh[3];
#pragma unroll
  for (int c = 0; c < 3; ++c) {
    const int j = t + c * 256;
    float a = b_ih[j], a2 = b_hh[j];
    for (int d = 0; d < D_SLOT; ++d) {
      a  += sx[d] * W_ih[(size_t)d * 768 + j];
      a2 += sh[d] * W_hh[(size_t)d * 768 + j];
    }
    gx[c] = a; gh[c] = a2;
  }
  const float r = 1.f / (1.f + __expf(-(gx[0] + gh[0])));
  const float z = 1.f / (1.f + __expf(-(gx[1] + gh[1])));
  const float n = tanhf(gx[2] + r * gh[2]);
  const float hnew = (1.f - z) * n + z * sh[t];
  shn[t] = hnew;

  float s = hnew, s2 = hnew * hnew;
#pragma unroll
  for (int m = 1; m < 32; m <<= 1) {
    s  += __shfl_xor(s,  m, 32);
    s2 += __shfl_xor(s2, m, 32);
  }
  if ((t & 31) == 0) { red[t >> 5] = s; red[8 + (t >> 5)] = s2; }
  __syncthreads();
  if (t == 0) {
    float ts = 0.f, ts2 = 0.f;
    for (int w = 0; w < 8; ++w) { ts += red[w]; ts2 += red[8 + w]; }
    const float mean = ts * (1.f / 256.f);
    const float var  = ts2 * (1.f / 256.f) - mean * mean;
    red[0] = mean; red[1] = rsqrtf(var + 1e-5f);
  }
  __syncthreads();
  ssn[t] = (hnew - red[0]) * red[1] * ln_g[t] + ln_b[t];
  __syncthreads();
#pragma unroll
  for (int c = 0; c < 2; ++c) {
    const int j = t + c * 256;
    float a = bm1[j];
    for (int d = 0; d < D_SLOT; ++d) a += ssn[d] * Wm1[(size_t)d * 512 + j];
    shid[j] = 0.5f * a * (1.f + erff(a * 0.70710678118654752f));
  }
  __syncthreads();
  float o = bm2[t];
  for (int d = 0; d < 512; ++d) o += shid[d] * Wm2[(size_t)d * D_SLOT + t];
  slots_out[(size_t)row * D_SLOT + t] = shn[t] + o;
}

// ---------------------------------------------------------------------------
// Pack outputs: slots, slots[:,:,:128], slots[:,:,128:]  (attn already placed)
// ---------------------------------------------------------------------------
__global__ __launch_bounds__(256) void k_pack(
    const float* __restrict__ slots, float* __restrict__ out)
{
  const int i = blockIdx.x * 256 + threadIdx.x;     // 65536 total
  const float v = slots[i];
  out[i] = v;
  const int bk = i >> 8, d = i & 255;
  const int base = 65536 + 1048576;                 // after slots + attn
  if (d < 128) out[base + bk * 128 + d] = v;
  else         out[base + 32768 + bk * 128 + (d - 128)] = v;
}

// ---------------------------------------------------------------------------
extern "C" void kernel_launch(void* const* d_in, const int* in_sizes, int n_in,
                              void* d_out, int out_size, void* d_ws, size_t ws_size,
                              hipStream_t stream) {
  (void)in_sizes; (void)n_in; (void)out_size; (void)ws_size;
  const float* features = (const float*)d_in[0];
  const float* Wi1 = (const float*)d_in[1];
  const float* bi1 = (const float*)d_in[2];
  const float* Wi2 = (const float*)d_in[3];
  const float* bi2 = (const float*)d_in[4];
  const float* Wp  = (const float*)d_in[5];
  const float* bp  = (const float*)d_in[6];
  const float* ln_in_g = (const float*)d_in[7];
  const float* ln_in_b = (const float*)d_in[8];
  const float* Wqs = (const float*)d_in[9];
  const float* Wks = (const float*)d_in[10];
  const float* Wqt = (const float*)d_in[11];
  const float* Wkt = (const float*)d_in[12];
  const float* W_ih = (const float*)d_in[13];
  const float* W_hh = (const float*)d_in[14];
  const float* b_ih = (const float*)d_in[15];
  const float* b_hh = (const float*)d_in[16];
  const float* ln_g = (const float*)d_in[17];
  const float* ln_b = (const float*)d_in[18];
  const float* Wm1 = (const float*)d_in[19];
  const float* bm1 = (const float*)d_in[20];
  const float* Wm2 = (const float*)d_in[21];
  const float* bm2 = (const float*)d_in[22];
  float* out = (float*)d_out;

  float* ws = (float*)d_ws;
  float* w_inputs = ws;   ws += (size_t)B_SZ * N_TOK * D_SLOT;        // 128 MB
  float* w_ks     = ws;   ws += (size_t)B_SZ * N_TOK * D_SH;          // 64 MB
  float* w_kt     = ws;   ws += (size_t)B_SZ * N_TOK * D_SH;          // 64 MB
  float* w_attn   = ws;   ws += (size_t)B_SZ * K_SLOT * N_TOK;        // 4 MB
  float* w_featp  = ws;   ws += (size_t)(B_SZ * N_TOK / MT) * D_IN;   // 12 MB
  float* w_fpool  = ws;   ws += B_SZ * D_IN;
  float* w_slotsA = ws;   ws += B_SZ * K_SLOT * D_SLOT;
  float* w_slotsB = ws;   ws += B_SZ * K_SLOT * D_SLOT;
  float* w_qs     = ws;   ws += B_SZ * 16 * D_SH;
  float* w_qt     = ws;   ws += B_SZ * 16 * D_SH;
  float* w_updp   = ws;   ws += B_SZ * 8 * K_SLOT * D_SLOT;
  float* w_upd    = ws;   ws += B_SZ * K_SLOT * D_SLOT;
  float* w_rspart = ws;   ws += B_SZ * 16 * K_SLOT;
  float* w_rowsum = ws;   ws += B_SZ * K_SLOT;

  k_proj_ln_keys<<<(B_SZ * N_TOK) / MT, 256, 0, stream>>>(
      features, Wp, bp, ln_in_g, ln_in_b, Wks, Wkt,
      w_inputs, w_ks, w_kt, w_featp);
  k_pool_reduce<<<B_SZ, 256, 0, stream>>>(w_featp, w_fpool);
  k_init_slots<<<B_SZ, 256, 0, stream>>>(w_fpool, Wi1, bi1, Wi2, bi2, w_slotsA);

  float* cur = w_slotsA;
  float* nxt = w_slotsB;
  for (int it = 0; it < 3; ++it) {
    float* attn_ptr = (it == 2) ? (out + B_SZ * K_SLOT * D_SLOT) : w_attn;
    k_qproj<<<B_SZ, 256, 0, stream>>>(cur, Wqs, Wqt, w_qs, w_qt);
    k_attn<<<dim3(N_TOK / 256, B_SZ), 256, 0, stream>>>(
        w_qs, w_qt, w_ks, w_kt, attn_ptr, w_rspart);
    k_rowsum_reduce<<<1, 256, 0, stream>>>(w_rspart, w_rowsum);
    k_updates<<<dim3(B_SZ, 8), 256, 0, stream>>>(attn_ptr, w_rowsum, w_inputs, w_updp);
    k_upd_reduce<<<256, 256, 0, stream>>>(w_updp, w_upd);
    k_gru_mlp<<<B_SZ * K_SLOT, 256, 0, stream>>>(
        w_upd, cur, W_ih, W_hh, b_ih, b_hh, ln_g, ln_b, Wm1, bm1, Wm2, bm2, nxt);
    float* tmp = cur; cur = nxt; nxt = tmp;
  }
  k_pack<<<(B_SZ * K_SLOT * D_SLOT) / 256, 256, 0, stream>>>(cur, out);
}
